// Attention_71683004171117
// MI455X (gfx1250) — compile-verified, matrix operand order checked by reading, not run on hardware
//
#include <hip/hip_runtime.h>

// ---- CDNA5 WMMA types ----
typedef __attribute__((ext_vector_type(16))) __bf16 v16bf;
typedef __attribute__((ext_vector_type(8)))  __bf16 v8bf;
typedef __attribute__((ext_vector_type(8)))  float  v8f;
typedef __attribute__((ext_vector_type(4)))  float  v4f;
typedef __attribute__((ext_vector_type(4)))  int    v4i;

struct BF8x2 { v8bf lo, hi; };
struct US16  { unsigned short s[16]; };

// gfx1250 async global->LDS copy path (ASYNCcnt), with sync fallback
#if defined(__has_builtin)
# if __has_builtin(__builtin_amdgcn_global_load_async_to_lds_b128) && \
     __has_builtin(__builtin_amdgcn_s_wait_asynccnt)
#  define HAVE_ASYNC 1
# endif
#endif
#ifndef HAVE_ASYNC
# define HAVE_ASYNC 0
#endif

static __device__ __forceinline__ void cp16_async(const float* g, float* l) {
#if HAVE_ASYNC
    // signature (from hipcc diagnostic): (v4i AS1*, v4i AS3*, imm offset, imm cpol)
    __builtin_amdgcn_global_load_async_to_lds_b128(
        (__attribute__((address_space(1))) v4i*)(void*)g,
        (__attribute__((address_space(3))) v4i*)(void*)l,
        0, 0);
#else
    *(v4f*)l = *(const v4f*)g;
#endif
}

static __device__ __forceinline__ void wait_async() {
#if HAVE_ASYNC
    __builtin_amdgcn_s_wait_asynccnt(0);
#endif
}

// float -> bf16 bits, round-to-nearest-even
static __device__ __forceinline__ unsigned short f2bfu(float f) {
    unsigned u = __float_as_uint(f);
    u += 0x7FFFu + ((u >> 16) & 1u);
    return (unsigned short)(u >> 16);
}

// Load one 16x16x32 A/B fragment (16 bf16 per lane) from a 64-wide LDS row.
// ISA 7.12.2 16-bit A layout: lane-half picks K base (0 or 8), chunks are
// [off, off+8) and [off+16, off+24) -> two 16B contiguous ds_load_b128.
static __device__ __forceinline__ v16bf ldsFrag(const unsigned short* rowPtr, int off) {
    BF8x2 t;
    t.lo = *(const v8bf*)(rowPtr + off);
    t.hi = *(const v8bf*)(rowPtr + off + 16);
    return __builtin_bit_cast(v16bf, t);
}

#define S_LEN 2048
#define D_LEN 64
#define TILE_F 4096   // 64x64 floats per K or V tile

__global__ __launch_bounds__(128)
void fa_causal_bf16_wmma(const float* __restrict__ Kg,
                         const float* __restrict__ Qg,
                         const float* __restrict__ Vg,
                         float* __restrict__ Og) {
    const int qb   = blockIdx.x;      // query block of 64 rows
    const int bh   = blockIdx.y;      // fused batch*head
    const int tid  = threadIdx.x;
    const int lane = tid & 31;
    const int wave = tid >> 5;        // 4 waves, 16 query rows each
    const int half = lane >> 4;       // 0/1: which K-half this lane holds
    const int ln   = lane & 15;

    // double-buffered raw f32 staging fed by async DMA: [buf][K | V]
    __shared__ float          Stg[2][2 * TILE_F];
    __shared__ unsigned short Ks[64][72];       // K tile bf16, row-major [key][d]
    __shared__ unsigned short Vs[64][72];       // V tile bf16, transposed [d][key]
    __shared__ unsigned short Ps[4][16][72];    // per-wave P scratch [row][key]

    const size_t base = (size_t)bh * S_LEN * D_LEN;
    const int qbase = qb * 64 + wave * 16;      // first query row of this wave

    // ---- Q fragments (A operand), scale folded in, loaded once ----
    const float scale = 0.125f;                 // 1/sqrt(64)
    const float* qrow = Qg + base + (size_t)(qbase + ln) * D_LEN;
    v16bf qf[2];
    #pragma unroll
    for (int c = 0; c < 2; ++c) {
        const int k0 = half * 8 + c * 32;
        US16 t;
        #pragma unroll
        for (int i = 0; i < 8; ++i) {
            t.s[i]     = f2bfu(qrow[k0 + i]      * scale);
            t.s[i + 8] = f2bfu(qrow[k0 + 16 + i] * scale);
        }
        qf[c] = __builtin_bit_cast(v16bf, t);
    }

    // ---- accumulators + online-softmax state ----
    v8f o[4];
    v8f zero = {0.f,0.f,0.f,0.f,0.f,0.f,0.f,0.f};
    #pragma unroll
    for (int t = 0; t < 4; ++t) o[t] = zero;
    float m[8], l[8];
    #pragma unroll
    for (int j = 0; j < 8; ++j) { m[j] = -1e30f; l[j] = 0.f; }

    // ---- prologue: start async DMA of KV tile 0 into staging buffer 0 ----
    {
        const float* kgp = Kg + base;
        const float* vgp = Vg + base;
        #pragma unroll
        for (int it = 0; it < 8; ++it) {
            const int off = it * 512 + tid * 4;
            cp16_async(kgp + off, &Stg[0][off]);
            cp16_async(vgp + off, &Stg[0][TILE_F + off]);
        }
    }

    // ---- causal flash loop over KV blocks (software-pipelined) ----
    for (int kb = 0; kb <= qb; ++kb) {
        const int cur = kb & 1;
        wait_async();        // this wave's DMA into Stg[cur] done
        __syncthreads();     // all waves' DMA done; prev compute reads done

        // kick off DMA for the next tile while we work on this one
        if (kb < qb) {
            const float* kgp = Kg + base + (size_t)(kb + 1) * TILE_F;
            const float* vgp = Vg + base + (size_t)(kb + 1) * TILE_F;
            #pragma unroll
            for (int it = 0; it < 8; ++it) {
                const int off = it * 512 + tid * 4;
                cp16_async(kgp + off, &Stg[1 - cur][off]);
                cp16_async(vgp + off, &Stg[1 - cur][TILE_F + off]);
            }
        }

        // convert staged f32 tile -> bf16 WMMA tiles (V transposed on store)
        #pragma unroll
        for (int it = 0; it < 8; ++it) {
            const int off = it * 512 + tid * 4;
            const int r = off >> 6, c = off & 63;
            v4f kd = *(const v4f*)(&Stg[cur][off]);
            v4f vd = *(const v4f*)(&Stg[cur][TILE_F + off]);
            #pragma unroll
            for (int i = 0; i < 4; ++i) {
                Ks[r][c + i] = f2bfu(kd[i]);
                Vs[c + i][r] = f2bfu(vd[i]);
            }
        }
        __syncthreads();

        // ---- scores S = (Q*scale) @ K^T : 16x64, 8 WMMAs ----
        v8f s[4];
        #pragma unroll
        for (int t = 0; t < 4; ++t) s[t] = zero;
        #pragma unroll
        for (int c = 0; c < 2; ++c) {
            #pragma unroll
            for (int t = 0; t < 4; ++t) {
                v16bf bf = ldsFrag(&Ks[t * 16 + ln][0], half * 8 + c * 32);
                s[t] = __builtin_amdgcn_wmma_f32_16x16x32_bf16(
                           false, qf[c], false, bf, (short)0, s[t], false, false);
            }
        }

        // ---- causal mask (only the diagonal block needs it) ----
        if (kb == qb) {
            #pragma unroll
            for (int t = 0; t < 4; ++t) {
                const int col = kb * 64 + t * 16 + ln;
                #pragma unroll
                for (int j = 0; j < 8; ++j) {
                    const int row = qbase + j + half * 8;
                    if (col > row) s[t][j] = -1e30f;
                }
            }
        }

        // ---- online softmax; C-layout row j lives across 16 lanes of a half ----
        #pragma unroll
        for (int j = 0; j < 8; ++j) {
            float rm = fmaxf(fmaxf(s[0][j], s[1][j]), fmaxf(s[2][j], s[3][j]));
            rm = fmaxf(rm, __shfl_xor(rm, 1, 32));
            rm = fmaxf(rm, __shfl_xor(rm, 2, 32));
            rm = fmaxf(rm, __shfl_xor(rm, 4, 32));
            rm = fmaxf(rm, __shfl_xor(rm, 8, 32));
            const float mn    = fmaxf(m[j], rm);
            const float alpha = __expf(m[j] - mn);
            m[j] = mn;
            float rs = 0.f;
            #pragma unroll
            for (int t = 0; t < 4; ++t) {
                const float p = __expf(s[t][j] - mn);
                s[t][j] = p;
                rs += p;
            }
            rs += __shfl_xor(rs, 1, 32);
            rs += __shfl_xor(rs, 2, 32);
            rs += __shfl_xor(rs, 4, 32);
            rs += __shfl_xor(rs, 8, 32);
            l[j] = l[j] * alpha + rs;
            #pragma unroll
            for (int t = 0; t < 4; ++t) o[t][j] *= alpha;
        }

        // ---- re-stripe P from C-layout to A-layout via per-wave LDS ----
        #pragma unroll
        for (int t = 0; t < 4; ++t)
            #pragma unroll
            for (int j = 0; j < 8; ++j)
                Ps[wave][j + half * 8][t * 16 + ln] = f2bfu(s[t][j]);
        // intra-wave DS ops stay in order: store->load RAW is safe

        // ---- O += P @ V : 8 WMMAs (B fragments from transposed V tile) ----
        #pragma unroll
        for (int c = 0; c < 2; ++c) {
            v16bf pa = ldsFrag(&Ps[wave][ln][0], half * 8 + c * 32);
            #pragma unroll
            for (int t = 0; t < 4; ++t) {
                v16bf vb = ldsFrag(&Vs[t * 16 + ln][0], half * 8 + c * 32);
                o[t] = __builtin_amdgcn_wmma_f32_16x16x32_bf16(
                           false, pa, false, vb, (short)0, o[t], false, false);
            }
        }
    }

    // ---- normalize and store (f32 output, [B,H,S,D]) ----
    float inv[8];
    #pragma unroll
    for (int j = 0; j < 8; ++j) inv[j] = 1.0f / l[j];
    #pragma unroll
    for (int t = 0; t < 4; ++t)
        #pragma unroll
        for (int j = 0; j < 8; ++j) {
            const int row = qbase + j + half * 8;
            Og[base + (size_t)row * D_LEN + t * 16 + ln] = o[t][j] * inv[j];
        }
}

extern "C" void kernel_launch(void* const* d_in, const int* in_sizes, int n_in,
                              void* d_out, int out_size, void* d_ws, size_t ws_size,
                              hipStream_t stream) {
    // setup_inputs order: k, q, v, mask (mask ignored: causality is structural)
    const float* k = (const float*)d_in[0];
    const float* q = (const float*)d_in[1];
    const float* v = (const float*)d_in[2];
    float* o = (float*)d_out;
    dim3 grid(S_LEN / 64, 32);   // 32 query blocks x (B*H = 32)
    fa_causal_bf16_wmma<<<grid, 128, 0, stream>>>(k, q, v, o);
}
